// VectorQuantizer_46892452938457
// MI455X (gfx1250) — matrix operand order repaired and checked by reading, not compiled, not run on hardware
//
#include <hip/hip_runtime.h>
#include <hip/hip_bf16.h>
#include <stddef.h>
#include <stdint.h>

typedef __bf16 bf16_t;
typedef __attribute__((ext_vector_type(16))) __bf16 v16bf;
typedef __attribute__((ext_vector_type(8)))  __bf16 v8bf;
typedef __attribute__((ext_vector_type(8)))  float  v8f;

#define NUM_EMB    8192
#define EMB_DIM    256
#define NROWS      16384          // 16 * 32 * 32
#define ZQ_ELEMS   4194304        // 16 * 256 * 32 * 32
#define LDS_STRIDE 264            // 256 + 8 bf16 pad -> conflict-free b128 loads

// workspace layout (bytes, all 256-aligned), total ~24.1 MB
#define OFF_XH  0u                // 16384*256 bf16  (8 MB)
#define OFF_XL  8388608u          // 8 MB
#define OFF_EH  16777216u         // 8192*256 bf16   (4 MB)
#define OFF_EL  20971520u         // 4 MB
#define OFF_EN  25165824u         // 8192 f32
#define OFF_IDX 25198592u         // 16384 i32
#define OFF_ACC 25264128u         // 1 f32 loss accumulator

// ---------------------------------------------------------------------------
// 1) hidden (B,C,H,W) f32 -> X (N,C) bf16 hi/lo split; zero loss accumulator
// ---------------------------------------------------------------------------
__global__ void vq_prep_x(const float* __restrict__ hidden,
                          bf16_t* __restrict__ Xh, bf16_t* __restrict__ Xl,
                          float* __restrict__ lossAcc) {
  if (blockIdx.x == 0 && threadIdx.x == 0) *lossAcc = 0.0f;
  const int bi = blockIdx.x;        // 16384 blocks = 16(b) * 32(y) * 32(c-group)
  const int cg = bi & 31;           // group of 8 channels
  const int y  = (bi >> 5) & 31;
  const int b  = bi >> 10;
  const int t  = threadIdx.x;       // 256
  const int tc = t & 7;
  const int tx = t >> 3;            // 0..31
  const int c  = cg * 8 + tc;
  const float v = hidden[(size_t)(b * 256 + c) * 1024 + y * 32 + tx];
  const bf16_t hi = (bf16_t)v;
  const bf16_t lo = (bf16_t)(v - (float)hi);
  const size_t dst = (size_t)(b * 1024 + y * 32 + tx) * 256 + c;
  Xh[dst] = hi;
  Xl[dst] = lo;
}

// ---------------------------------------------------------------------------
// 2) codebook f32 -> bf16 hi/lo split + per-code squared L2 norm
// ---------------------------------------------------------------------------
__global__ void vq_prep_e(const float* __restrict__ emb,
                          bf16_t* __restrict__ Eh, bf16_t* __restrict__ El,
                          float* __restrict__ enorm) {
  __shared__ float red[256];
  const int k = blockIdx.x;
  const int c = threadIdx.x;
  const float v = emb[(size_t)k * 256 + c];
  const bf16_t hi = (bf16_t)v;
  const bf16_t lo = (bf16_t)(v - (float)hi);
  const size_t o = (size_t)k * 256 + c;
  Eh[o] = hi;
  El[o] = lo;
  red[c] = v * v;
  __syncthreads();
  for (int s = 128; s > 0; s >>= 1) {
    if (c < s) red[c] += red[c + s];
    __syncthreads();
  }
  if (c == 0) enorm[k] = red[0];
}

// ---------------------------------------------------------------------------
// async global -> LDS fill of one 32-code tile (hi + lo), no VGPR bounce.
// Each thread issues 8x global_load_async_to_lds_b128 (tracked by ASYNCcnt).
// ---------------------------------------------------------------------------
__device__ __forceinline__ void async_fill_tile(const bf16_t* __restrict__ Eh,
                                                const bf16_t* __restrict__ El,
                                                bf16_t* ldsEh, bf16_t* ldsEl,
                                                int codebase, int tid) {
#pragma unroll
  for (int i = 0; i < 4; ++i) {
    const int chunk = tid + i * 256;       // 1024 chunks of 8 bf16 per array
    const int r  = chunk >> 5;
    const int cc = (chunk & 31) * 8;
    const bf16_t* gh = Eh + (size_t)(codebase + r) * 256 + cc;
    const bf16_t* gl = El + (size_t)(codebase + r) * 256 + cc;
    const uint32_t dh = (uint32_t)(uintptr_t)(ldsEh + r * LDS_STRIDE + cc);
    const uint32_t dl = (uint32_t)(uintptr_t)(ldsEl + r * LDS_STRIDE + cc);
    asm volatile("global_load_async_to_lds_b128 %0, %1, off"
                 :: "v"(dh), "v"(gh) : "memory");
    asm volatile("global_load_async_to_lds_b128 %0, %1, off"
                 :: "v"(dl), "v"(gl) : "memory");
  }
}

// ---------------------------------------------------------------------------
// 3) fused WMMA GEMM (X @ E^T) + running argmin of (||e||^2 - 2 x.e)
//    8 waves/block, 16 rows/wave, A fragments resident in VGPRs.
//    E tiles double-buffered in LDS via async tensor-path copies so the
//    fill of tile k+1 overlaps the WMMA work on tile k.
// ---------------------------------------------------------------------------
__global__ __launch_bounds__(256) void vq_gemm_argmin(
    const bf16_t* __restrict__ Xh, const bf16_t* __restrict__ Xl,
    const bf16_t* __restrict__ Eh, const bf16_t* __restrict__ El,
    const float* __restrict__ enorm, int* __restrict__ idxOut) {
  __shared__ bf16_t ldsEh[2][32 * LDS_STRIDE];
  __shared__ bf16_t ldsEl[2][32 * LDS_STRIDE];

  const int tid  = threadIdx.x;
  const int lane = tid & 31;
  const int wave = tid >> 5;
  const int m    = lane & 15;       // A row / B column within tile
  const int half = lane >> 4;       // K half-selector for 16-bit fragments
  const int rowbase = blockIdx.x * 128 + wave * 16;
  const int arow    = rowbase + m;

  // A fragments: 16 rows x K=256, hi + lo, kept in registers for entire scan.
  // 16-bit A layout: lanes 0-15 hold K {0..7,16..23}, lanes 16-31 K {8..15,24..31}.
  v16bf Ah[8], Al[8];
#pragma unroll
  for (int ch = 0; ch < 8; ++ch) {
    const bf16_t* p = Xh + (size_t)arow * 256 + ch * 32 + half * 8;
    ((v8bf*)&Ah[ch])[0] = *(const v8bf*)(p);
    ((v8bf*)&Ah[ch])[1] = *(const v8bf*)(p + 16);
    const bf16_t* q = Xl + (size_t)arow * 256 + ch * 32 + half * 8;
    ((v8bf*)&Al[ch])[0] = *(const v8bf*)(q);
    ((v8bf*)&Al[ch])[1] = *(const v8bf*)(q + 16);
  }

  float minv[8];
  int   mini[8];
#pragma unroll
  for (int v = 0; v < 8; ++v) { minv[v] = 3.4e38f; mini[v] = 0; }

  // prologue: start fill of tile 0 into buffer 0
  async_fill_tile(Eh, El, ldsEh[0], ldsEl[0], 0, tid);

  for (int kt = 0; kt < NUM_EMB / 32; ++kt) {
    const int cur      = kt & 1;
    const int codebase = kt * 32;

    // my async fills have landed; barrier publishes everyone's
    asm volatile("s_wait_asynccnt 0x0" ::: "memory");
    __syncthreads();

    // kick off the next tile's fill into the other buffer (overlaps compute)
    if (kt + 1 < NUM_EMB / 32)
      async_fill_tile(Eh, El, ldsEh[cur ^ 1], ldsEl[cur ^ 1],
                      codebase + 32, tid);

    const bf16_t* bufH = ldsEh[cur];
    const bf16_t* bufL = ldsEl[cur];

#pragma unroll
    for (int sub = 0; sub < 2; ++sub) {
      const int codeb = codebase + sub * 16;
      const int brow  = sub * 16 + m;
      v8f acc = {};
#pragma unroll
      for (int ch = 0; ch < 8; ++ch) {
        // 16-bit B layout: lanes 0-15 hold K {0..7,16..23}, lanes 16-31 K {8..15,24..31}
        const bf16_t* pb = &bufH[brow * LDS_STRIDE + ch * 32 + half * 8];
        v16bf Bh;
        ((v8bf*)&Bh)[0] = *(const v8bf*)(pb);
        ((v8bf*)&Bh)[1] = *(const v8bf*)(pb + 16);
        const bf16_t* pc = &bufL[brow * LDS_STRIDE + ch * 32 + half * 8];
        v16bf Bl;
        ((v8bf*)&Bl)[0] = *(const v8bf*)(pc);
        ((v8bf*)&Bl)[1] = *(const v8bf*)(pc + 16);
        // split-bf16 compensated fp32 dot: hi*hi + lo*hi + hi*lo
        acc = __builtin_amdgcn_wmma_f32_16x16x32_bf16(false, Ah[ch], false, Bh,
                                                      (short)0, acc, false, false);
        acc = __builtin_amdgcn_wmma_f32_16x16x32_bf16(false, Al[ch], false, Bh,
                                                      (short)0, acc, false, false);
        acc = __builtin_amdgcn_wmma_f32_16x16x32_bf16(false, Ah[ch], false, Bl,
                                                      (short)0, acc, false, false);
      }
      const float en   = enorm[codeb + m];
      const int   code = codeb + m;
#pragma unroll
      for (int v = 0; v < 8; ++v) {
        // D layout: lanes 0-15 -> row v, lanes 16-31 -> row v+8, col = lane&15
        const float d = en - 2.0f * acc[v];
        if (d < minv[v]) { minv[v] = d; mini[v] = code; }
      }
    }
    // all readers of this buffer done before it is refilled two tiles later
    __syncthreads();
  }

  // (min, argmin) reduction across the 16 lanes sharing each output row.
  // xor masks 1,2,4,8 stay inside each half-wave; tie -> lower code index.
#pragma unroll
  for (int v = 0; v < 8; ++v) {
    float mv = minv[v];
    int   mi = mini[v];
#pragma unroll
    for (int off = 1; off < 16; off <<= 1) {
      const float ov = __shfl_xor(mv, off, 32);
      const int   oi = __shfl_xor(mi, off, 32);
      if (ov < mv || (ov == mv && oi < mi)) { mv = ov; mi = oi; }
    }
    if (m == 0) idxOut[rowbase + half * 8 + v] = mi;
  }
}

// ---------------------------------------------------------------------------
// 4) gather z_q = emb[idx] (fp32 exact), write (B,C,H,W), accumulate loss,
//    emit indices as floats into d_out tail.
// ---------------------------------------------------------------------------
__global__ void vq_gather_loss(const float* __restrict__ hidden,
                               const float* __restrict__ emb,
                               const int* __restrict__ idxIn,
                               float* __restrict__ out,
                               float* __restrict__ lossAcc) {
  __shared__ float red[256];
  const int n = blockIdx.x;
  const int c = threadIdx.x;
  const int b = n >> 10, rem = n & 1023, y = rem >> 5, x = rem & 31;
  const int idx = idxIn[n];
  const float e = emb[(size_t)idx * 256 + c];
  const size_t hoff = (size_t)(b * 256 + c) * 1024 + y * 32 + x;
  const float h = hidden[hoff];
  out[hoff] = e;                     // straight-through forward value == z_q
  const float d = e - h;
  red[c] = d * d;
  __syncthreads();
  for (int s = 128; s > 0; s >>= 1) {
    if (c < s) red[c] += red[c + s];
    __syncthreads();
  }
  if (c == 0) {
    atomicAdd(lossAcc, red[0]);
    out[ZQ_ELEMS + n] = (float)idx;  // indices block of d_out
  }
}

// ---------------------------------------------------------------------------
// 5) loss = mean((sg(zq)-h)^2) + 0.25*mean((zq-sg(h))^2) = 1.25*mean(diff^2)
// ---------------------------------------------------------------------------
__global__ void vq_finalize(const float* __restrict__ lossAcc,
                            float* __restrict__ out) {
  out[ZQ_ELEMS + NROWS] = 1.25f * (*lossAcc) / (float)ZQ_ELEMS;
}

extern "C" void kernel_launch(void* const* d_in, const int* in_sizes, int n_in,
                              void* d_out, int out_size, void* d_ws, size_t ws_size,
                              hipStream_t stream) {
  (void)in_sizes; (void)n_in; (void)out_size; (void)ws_size;
  const float* hidden = (const float*)d_in[0];
  const float* emb    = (const float*)d_in[1];
  float* out = (float*)d_out;
  char*  ws  = (char*)d_ws;
  bf16_t* Xh  = (bf16_t*)(ws + OFF_XH);
  bf16_t* Xl  = (bf16_t*)(ws + OFF_XL);
  bf16_t* Eh  = (bf16_t*)(ws + OFF_EH);
  bf16_t* El  = (bf16_t*)(ws + OFF_EL);
  float*  en  = (float*) (ws + OFF_EN);
  int*    idx = (int*)   (ws + OFF_IDX);
  float*  acc = (float*) (ws + OFF_ACC);

  vq_prep_x     <<<16384, 256, 0, stream>>>(hidden, Xh, Xl, acc);
  vq_prep_e     <<<8192,  256, 0, stream>>>(emb, Eh, El, en);
  vq_gemm_argmin<<<128,   256, 0, stream>>>(Xh, Xl, Eh, El, en, idx);
  vq_gather_loss<<<16384, 256, 0, stream>>>(hidden, emb, idx, out, acc);
  vq_finalize   <<<1,     1,   0, stream>>>(acc, out);
}
